// MultiModalEmbedder_67671504716322
// MI455X (gfx1250) — compile-verified, matrix operand order checked by reading, not run on hardware
//
#include <hip/hip_runtime.h>
#include <hip/hip_bf16.h>
#include <math.h>

// ---------------------------------------------------------------------------
// MultiModalEmbedder forward for MI455X (gfx1250, wave32, WMMA).
// Dense layers: v_wmma_f32_16x16x32_f16 (fp16 in, fp32 accum).
// Each wave owns a 32x64 output strip: 2 A-frags x 4 B-frags -> 8 WMMA/k-step.
// K-loop kept at unroll 1 so accumulators stay in VGPRs (no scratch spill).
// Graph attention: 3-pass segment softmax with f32 global atomics.
// ---------------------------------------------------------------------------

#define NN 50000      // nodes
#define NE 400000     // edges
#define HEADS 2
#define HDIM 256

typedef __attribute__((ext_vector_type(16))) _Float16 v16h;
typedef __attribute__((ext_vector_type(8)))  _Float16 v8h;
typedef __attribute__((ext_vector_type(8)))  float    v8f;

union AFrag { v16h v; v8h h[2]; };

enum { ACT_NONE = 0, ACT_LEAKY02, ACT_RELU, ACT_GELU, ACT_ELU, ACT_LEAKY001 };

__device__ __forceinline__ float apply_act(float x, int act) {
  switch (act) {
    case ACT_LEAKY02:  return x > 0.f ? x : 0.2f * x;
    case ACT_RELU:     return x > 0.f ? x : 0.f;
    case ACT_GELU:     return 0.5f * x * (1.f + erff(x * 0.70710678118654752f));
    case ACT_ELU:      return x > 0.f ? x : (__expf(x) - 1.f);
    case ACT_LEAKY001: return x > 0.f ? x : 0.01f * x;
    default:           return x;
  }
}

// ---------------------------------------------------------------------------
// WMMA GEMM: C[M,N] = act(A[M,K]_f16 * W^T[N,K]_f16 + bias).
// Wave strip: 32 rows (2 M-tiles, rows clamped to M-1 on load, guarded on
// store) x 64 cols (4 N-tiles, cols clamped to N-1 on load, guarded on store).
// ---------------------------------------------------------------------------
__global__ __launch_bounds__(256, 1)
void gemm_wmma_f16(const _Float16* __restrict__ A, long lda,
                   const _Float16* __restrict__ Bt, long ldb,
                   const float* __restrict__ bias,
                   float* __restrict__ outF, long ldoF,
                   _Float16* __restrict__ outH, long ldoH,
                   int M, int N, int K, int act) {
  const int waveId = (int)((blockIdx.x * (long)blockDim.x + threadIdx.x) >> 5);
  const int lane   = threadIdx.x & 31;
  const int nGroups = (N + 63) >> 6;
  const int mTiles  = (M + 31) >> 5;
  const int mt = waveId / nGroups;
  const int ng = waveId % nGroups;
  if (mt >= mTiles) return;

  const int lane15 = lane & 15;
  const int kb     = (lane >> 4) << 3;   // per-lane K sub-offset: 0 or 8
  const int n0     = ng << 6;

  // Clamped A row pointers (2 M-tiles of 16 rows each).
  int row0 = mt * 32 + lane15;           if (row0 > M - 1) row0 = M - 1;
  int row1 = mt * 32 + 16 + lane15;      if (row1 > M - 1) row1 = M - 1;
  const _Float16* a0p = A + (size_t)row0 * lda + kb;
  const _Float16* a1p = A + (size_t)row1 * lda + kb;

  // Clamped B column pointers (4 N-tiles of 16 cols each).
  int c0 = n0 + lane15;       if (c0 > N - 1) c0 = N - 1;
  int c1 = n0 + 16 + lane15;  if (c1 > N - 1) c1 = N - 1;
  int c2 = n0 + 32 + lane15;  if (c2 > N - 1) c2 = N - 1;
  int c3 = n0 + 48 + lane15;  if (c3 > N - 1) c3 = N - 1;
  const _Float16* b0p = Bt + (size_t)c0 * ldb + kb;
  const _Float16* b1p = Bt + (size_t)c1 * ldb + kb;
  const _Float16* b2p = Bt + (size_t)c2 * ldb + kb;
  const _Float16* b3p = Bt + (size_t)c3 * ldb + kb;

  v8f acc[2][4] = {};

#pragma unroll 1
  for (int kk = K >> 5; kk > 0; --kk) {
    AFrag a0, a1, b0, b1, b2, b3;
    a0.h[0] = *(const v8h*)(a0p);
    a0.h[1] = *(const v8h*)(a0p + 16);
    a1.h[0] = *(const v8h*)(a1p);
    a1.h[1] = *(const v8h*)(a1p + 16);
    b0.h[0] = *(const v8h*)(b0p);
    b0.h[1] = *(const v8h*)(b0p + 16);
    b1.h[0] = *(const v8h*)(b1p);
    b1.h[1] = *(const v8h*)(b1p + 16);
    b2.h[0] = *(const v8h*)(b2p);
    b2.h[1] = *(const v8h*)(b2p + 16);
    b3.h[0] = *(const v8h*)(b3p);
    b3.h[1] = *(const v8h*)(b3p + 16);
    // Speculative prefetch 512B ahead (silently dropped if past the buffer).
    __builtin_prefetch(a0p + 256, 0, 3);
    __builtin_prefetch(a1p + 256, 0, 3);
    a0p += 32; a1p += 32;
    b0p += 32; b1p += 32; b2p += 32; b3p += 32;

    acc[0][0] = __builtin_amdgcn_wmma_f32_16x16x32_f16(false, a0.v, false, b0.v, (short)0, acc[0][0], false, false);
    acc[0][1] = __builtin_amdgcn_wmma_f32_16x16x32_f16(false, a0.v, false, b1.v, (short)0, acc[0][1], false, false);
    acc[0][2] = __builtin_amdgcn_wmma_f32_16x16x32_f16(false, a0.v, false, b2.v, (short)0, acc[0][2], false, false);
    acc[0][3] = __builtin_amdgcn_wmma_f32_16x16x32_f16(false, a0.v, false, b3.v, (short)0, acc[0][3], false, false);
    acc[1][0] = __builtin_amdgcn_wmma_f32_16x16x32_f16(false, a1.v, false, b0.v, (short)0, acc[1][0], false, false);
    acc[1][1] = __builtin_amdgcn_wmma_f32_16x16x32_f16(false, a1.v, false, b1.v, (short)0, acc[1][1], false, false);
    acc[1][2] = __builtin_amdgcn_wmma_f32_16x16x32_f16(false, a1.v, false, b2.v, (short)0, acc[1][2], false, false);
    acc[1][3] = __builtin_amdgcn_wmma_f32_16x16x32_f16(false, a1.v, false, b3.v, (short)0, acc[1][3], false, false);
  }

  // C/D layout: lane holds column (lane&15), rows (lane>=16 ? 8 : 0) + r.
  const int rsub = (lane >> 4) << 3;
#pragma unroll
  for (int mi = 0; mi < 2; ++mi) {
    const int rbase = mt * 32 + mi * 16 + rsub;
#pragma unroll
    for (int j = 0; j < 4; ++j) {
      int c = n0 + j * 16 + lane15;
      if (c >= N) continue;
      float bv = bias[c];
#pragma unroll
      for (int r = 0; r < 8; ++r) {
        int m = rbase + r;
        if (m >= M) continue;
        float x = apply_act(acc[mi][j][r] + bv, act);
        size_t rr = (size_t)m;
        if (outF) outF[rr * ldoF + c] = x;
        if (outH) outH[rr * ldoH + c] = (_Float16)x;
      }
    }
  }
}

// ---------------------------------------------------------------------------
// Weight convert: W[K,N] f32 row-major -> Wt[N,Kpad] f16 (zero-padded K).
// ---------------------------------------------------------------------------
__global__ void wconv_kernel(const float* __restrict__ W, _Float16* __restrict__ Wt,
                             int K, int N, int Kpad) {
  long idx = blockIdx.x * (long)blockDim.x + threadIdx.x;
  long total = (long)N * Kpad;
  if (idx >= total) return;
  int n = (int)(idx / Kpad), k = (int)(idx % Kpad);
  Wt[(size_t)n * Kpad + k] = (k < K) ? (_Float16)W[(size_t)k * N + n] : (_Float16)0.f;
}

// ---------------------------------------------------------------------------
// LayerNorm (f32 in) -> f16 out, zero-padded to Dpad columns. Block per row.
// ---------------------------------------------------------------------------
__global__ void layernorm_f16_kernel(const float* __restrict__ X,
                                     const float* __restrict__ g,
                                     const float* __restrict__ b,
                                     _Float16* __restrict__ out,
                                     int D, int Dpad, float invD) {
  const float* x = X + (size_t)blockIdx.x * D;
  float s = 0.f, s2 = 0.f;
  for (int i = threadIdx.x; i < D; i += blockDim.x) {
    float v = x[i]; s += v; s2 += v * v;
  }
  for (int off = 16; off; off >>= 1) { s += __shfl_down(s, off); s2 += __shfl_down(s2, off); }
  __shared__ float sS[8], sS2[8];
  int wid = threadIdx.x >> 5, lane = threadIdx.x & 31;
  if (lane == 0) { sS[wid] = s; sS2[wid] = s2; }
  __syncthreads();
  if (threadIdx.x == 0) {
    float ts = 0.f, ts2 = 0.f;
    int nw = blockDim.x >> 5;
    for (int w = 0; w < nw; ++w) { ts += sS[w]; ts2 += sS2[w]; }
    sS[0] = ts; sS2[0] = ts2;
  }
  __syncthreads();
  float mu  = sS[0] * invD;
  float var = sS2[0] * invD - mu * mu;
  float r   = rsqrtf(var + 1e-5f);
  _Float16* o = out + (size_t)blockIdx.x * Dpad;
  for (int i = threadIdx.x; i < Dpad; i += blockDim.x) {
    float y = (i < D) ? (x[i] - mu) * r * g[i] + b[i] : 0.f;
    o[i] = (_Float16)y;
  }
}

// ---------------------------------------------------------------------------
// Geo encoder (LN(2) -> 64 relu -> 32), scalar, weights cached in LDS.
// Writes f16 into combined[:, 256:288].
// ---------------------------------------------------------------------------
__global__ void geo_encoder_kernel(const float* __restrict__ geo,
                                   const float* __restrict__ lng, const float* __restrict__ lnb,
                                   const float* __restrict__ w1, const float* __restrict__ b1,
                                   const float* __restrict__ w2, const float* __restrict__ b2,
                                   _Float16* __restrict__ comb) {
  __shared__ float sw1[128], sb1[64], sw2[2048], sb2[32], sg[2], sb[2];
  for (int i = threadIdx.x; i < 128;  i += blockDim.x) sw1[i] = w1[i];
  for (int i = threadIdx.x; i < 64;   i += blockDim.x) sb1[i] = b1[i];
  for (int i = threadIdx.x; i < 2048; i += blockDim.x) sw2[i] = w2[i];
  for (int i = threadIdx.x; i < 32;   i += blockDim.x) sb2[i] = b2[i];
  if (threadIdx.x < 2) { sg[threadIdx.x] = lng[threadIdx.x]; sb[threadIdx.x] = lnb[threadIdx.x]; }
  __syncthreads();
  int n = blockIdx.x * blockDim.x + threadIdx.x;
  if (n >= NN) return;
  float x0 = geo[(size_t)n * 2], x1 = geo[(size_t)n * 2 + 1];
  float mu = 0.5f * (x0 + x1);
  float d0 = x0 - mu, d1 = x1 - mu;
  float r  = rsqrtf(0.5f * (d0 * d0 + d1 * d1) + 1e-5f);
  x0 = d0 * r * sg[0] + sb[0];
  x1 = d1 * r * sg[1] + sb[1];
  float h[64];
#pragma unroll
  for (int j = 0; j < 64; ++j) {
    float v = x0 * sw1[j] + x1 * sw1[64 + j] + sb1[j];
    h[j] = v > 0.f ? v : 0.f;
  }
  _Float16* o = comb + (size_t)n * 928 + 256;
  for (int c = 0; c < 32; ++c) {
    float acc = sb2[c];
#pragma unroll
    for (int j = 0; j < 64; ++j) acc += h[j] * sw2[j * 32 + c];
    o[c] = (_Float16)acc;
  }
}

// ---------------------------------------------------------------------------
// Attention: segment softmax over dst nodes.
// ---------------------------------------------------------------------------
__device__ __forceinline__ unsigned fenc(float f) {
  unsigned u = __float_as_uint(f);
  return (u & 0x80000000u) ? ~u : (u | 0x80000000u);
}
__device__ __forceinline__ float fdec(unsigned e) {
  return __uint_as_float((e & 0x80000000u) ? (e ^ 0x80000000u) : ~e);
}

__global__ void attn_init_kernel(unsigned* __restrict__ mbuf, float* __restrict__ denom) {
  int i = blockIdx.x * blockDim.x + threadIdx.x;
  if (i >= NN * HEADS) return;
  mbuf[i] = 0x007FFFFFu;  // fenc(-inf)
  denom[i] = 0.f;
}

// One wave per (edge, head): 256-length dot product + segment max.
__global__ void edge_score_kernel(const float* __restrict__ q, const float* __restrict__ k,
                                  const int* __restrict__ ei,
                                  float* __restrict__ score, unsigned* __restrict__ mbuf) {
  int gw = (int)((blockIdx.x * (long)blockDim.x + threadIdx.x) >> 5);
  int lane = threadIdx.x & 31;
  if (gw >= NE * HEADS) return;
  int e = gw >> 1, h = gw & 1;
  int src = ei[e], dst = ei[NE + e];
  const float* qp = q + (size_t)dst * 512 + h * 256;
  const float* kp = k + (size_t)src * 512 + h * 256;
  float s = 0.f;
  for (int i = lane; i < 256; i += 32) s += qp[i] * kp[i];
  for (int off = 16; off; off >>= 1) s += __shfl_down(s, off);
  if (lane == 0) {
    s *= 0.0625f;  // 1/sqrt(256)
    score[(size_t)e * 2 + h] = s;
    atomicMax(&mbuf[dst * 2 + h], fenc(s));
  }
}

__global__ void edge_exp_kernel(const int* __restrict__ ei,
                                const unsigned* __restrict__ mbuf,
                                float* __restrict__ score, float* __restrict__ denom) {
  int t = blockIdx.x * blockDim.x + threadIdx.x;
  if (t >= NE * HEADS) return;
  int e = t >> 1, h = t & 1;
  int dst = ei[NE + e];
  float ex = __expf(score[(size_t)e * 2 + h] - fdec(mbuf[dst * 2 + h]));
  score[(size_t)e * 2 + h] = ex;
  atomicAdd(&denom[dst * 2 + h], ex);
}

// One wave per (edge, head): scatter v[src]*alpha onto skip buffer at dst.
__global__ void edge_msg_kernel(const float* __restrict__ v, const int* __restrict__ ei,
                                const float* __restrict__ score, const float* __restrict__ denom,
                                float* __restrict__ outbuf) {
  int gw = (int)((blockIdx.x * (long)blockDim.x + threadIdx.x) >> 5);
  int lane = threadIdx.x & 31;
  if (gw >= NE * HEADS) return;
  int e = gw >> 1, h = gw & 1;
  int src = ei[e], dst = ei[NE + e];
  float alpha = score[(size_t)e * 2 + h] / (denom[dst * 2 + h] + 1e-16f);
  const float* vp = v + (size_t)src * 512 + h * 256;
  float* op = outbuf + (size_t)dst * 512 + h * 256;
  for (int i = lane; i < 256; i += 32) atomicAdd(&op[i], vp[i] * alpha);
}

// leaky_relu(0.01) of (skip + messages), fp16 into combined[:, 416:928].
__global__ void graph_epilogue_kernel(const float* __restrict__ sbuf, _Float16* __restrict__ comb) {
  long i = blockIdx.x * (long)blockDim.x + threadIdx.x;
  if (i >= (long)NN * 512) return;
  int n = (int)(i >> 9), c = (int)(i & 511);
  float x = sbuf[i];
  comb[(size_t)n * 928 + 416 + c] = (_Float16)(x > 0.f ? x : 0.01f * x);
}

// Geo decoder: rec_geo = emb @ do_w + do_b (256 -> 2), scalar.
__global__ void geo_decoder_kernel(const float* __restrict__ emb,
                                   const float* __restrict__ w, const float* __restrict__ b,
                                   float* __restrict__ out) {
  __shared__ float sw[512];
  for (int i = threadIdx.x; i < 512; i += blockDim.x) sw[i] = w[i];
  __syncthreads();
  int n = blockIdx.x * blockDim.x + threadIdx.x;
  if (n >= NN) return;
  const float* e = emb + (size_t)n * 256;
  float a0 = b[0], a1 = b[1];
  for (int i = 0; i < 256; ++i) {
    float v = e[i];
    a0 += v * sw[i * 2];
    a1 += v * sw[i * 2 + 1];
  }
  out[(size_t)n * 2]     = a0;
  out[(size_t)n * 2 + 1] = a1;
}

// ---------------------------------------------------------------------------
// Host side
// ---------------------------------------------------------------------------
static void launch_gemm(hipStream_t s, const _Float16* A, long lda,
                        const _Float16* Bt, long ldb, const float* bias,
                        float* outF, long ldoF, _Float16* outH, long ldoH,
                        int M, int N, int K, int act) {
  long waves = (long)((M + 31) / 32) * ((N + 63) / 64);
  long blocks = (waves * 32 + 255) / 256;
  gemm_wmma_f16<<<dim3((unsigned)blocks), dim3(256), 0, s>>>(
      A, lda, Bt, ldb, bias, outF, ldoF, outH, ldoH, M, N, K, act);
}

static void launch_wconv(hipStream_t s, const float* W, _Float16* Wt, int K, int N, int Kpad) {
  long total = (long)N * Kpad;
  wconv_kernel<<<dim3((unsigned)((total + 255) / 256)), dim3(256), 0, s>>>(W, Wt, K, N, Kpad);
}

extern "C" void kernel_launch(void* const* d_in, const int* in_sizes, int n_in,
                              void* d_out, int out_size, void* d_ws, size_t ws_size,
                              hipStream_t stream) {
  (void)in_sizes; (void)n_in; (void)out_size; (void)ws_size;

  // Inputs in setup_inputs() dict order.
  const float* gram = (const float*)d_in[0];
  const float* geo  = (const float*)d_in[1];
  const float* text = (const float*)d_in[2];
  const int*   ei   = (const int*)d_in[3];
  const float *g_ln_g = (const float*)d_in[4],  *g_ln_b = (const float*)d_in[5];
  const float *g_w1 = (const float*)d_in[6],  *g_b1 = (const float*)d_in[7];
  const float *g_w2 = (const float*)d_in[8],  *g_b2 = (const float*)d_in[9];
  const float *o_ln_g = (const float*)d_in[10], *o_ln_b = (const float*)d_in[11];
  const float *o_w1 = (const float*)d_in[12], *o_b1 = (const float*)d_in[13];
  const float *o_w2 = (const float*)d_in[14], *o_b2 = (const float*)d_in[15];
  const float *t_ln_g = (const float*)d_in[16], *t_ln_b = (const float*)d_in[17];
  const float *t_w1 = (const float*)d_in[18], *t_b1 = (const float*)d_in[19];
  const float *t_w2 = (const float*)d_in[20], *t_b2 = (const float*)d_in[21];
  const float *q_w = (const float*)d_in[22],  *q_b = (const float*)d_in[23];
  const float *k_w = (const float*)d_in[24],  *k_b = (const float*)d_in[25];
  const float *v_w = (const float*)d_in[26],  *v_b = (const float*)d_in[27];
  const float *s_w = (const float*)d_in[28],  *s_b = (const float*)d_in[29];
  const float *p_w1 = (const float*)d_in[30], *p_b1 = (const float*)d_in[31];
  const float *p_w2 = (const float*)d_in[32], *p_b2 = (const float*)d_in[33];
  const float *dg_w = (const float*)d_in[34], *dg_b = (const float*)d_in[35];
  const float *do_w = (const float*)d_in[36], *do_b = (const float*)d_in[37];
  const float *dt_w = (const float*)d_in[38], *dt_b = (const float*)d_in[39];

  // Output layout (floats): rec_gram | rec_geo | rec_text | emb
  float* out = (float*)d_out;
  float* rec_gram = out;
  float* rec_geo  = out + (size_t)NN * 400;
  float* rec_text = out + (size_t)NN * 402;
  float* emb      = out + (size_t)NN * 1170;

  // Workspace bump allocator (256B aligned).
  size_t off = 0;
  char* ws = (char*)d_ws;
  auto alloc = [&](size_t bytes) -> char* {
    char* p = ws + off;
    off += (bytes + 255) & ~(size_t)255;
    return p;
  };
  _Float16* g1t = (_Float16*)alloc((size_t)512 * 416 * 2);
  _Float16* g2t = (_Float16*)alloc((size_t)256 * 512 * 2);
  _Float16* t1t = (_Float16*)alloc((size_t)256 * 768 * 2);
  _Float16* t2t = (_Float16*)alloc((size_t)128 * 256 * 2);
  _Float16* qt  = (_Float16*)alloc((size_t)512 * 256 * 2);
  _Float16* kt  = (_Float16*)alloc((size_t)512 * 256 * 2);
  _Float16* vt  = (_Float16*)alloc((size_t)512 * 256 * 2);
  _Float16* st  = (_Float16*)alloc((size_t)512 * 256 * 2);
  _Float16* p1t = (_Float16*)alloc((size_t)1024 * 928 * 2);
  _Float16* p2t = (_Float16*)alloc((size_t)256 * 1024 * 2);
  _Float16* dgt = (_Float16*)alloc((size_t)400 * 256 * 2);
  _Float16* dtt = (_Float16*)alloc((size_t)768 * 256 * 2);

  _Float16* gln   = (_Float16*)alloc((size_t)NN * 416 * 2);
  _Float16* h512  = (_Float16*)alloc((size_t)NN * 512 * 2);
  _Float16* tln   = (_Float16*)alloc((size_t)NN * 768 * 2);
  _Float16* th    = (_Float16*)alloc((size_t)NN * 256 * 2);
  _Float16* comb  = (_Float16*)alloc((size_t)NN * 928 * 2);
  float*    qb    = (float*)alloc((size_t)NN * 512 * 4);
  float*    kb    = (float*)alloc((size_t)NN * 512 * 4);
  float*    vb    = (float*)alloc((size_t)NN * 512 * 4);
  float*    sbuf  = (float*)alloc((size_t)NN * 512 * 4);
  _Float16* h1024 = (_Float16*)alloc((size_t)NN * 1024 * 2);
  _Float16* embh  = (_Float16*)alloc((size_t)NN * 256 * 2);
  float*    score = (float*)alloc((size_t)NE * 2 * 4);
  unsigned* mbuf  = (unsigned*)alloc((size_t)NN * 2 * 4);
  float*    denom = (float*)alloc((size_t)NN * 2 * 4);

  // 1) Transpose + fp16-convert all GEMM weights (small; L2-resident after).
  launch_wconv(stream, g_w1, g1t, 400, 512, 416);
  launch_wconv(stream, g_w2, g2t, 512, 256, 512);
  launch_wconv(stream, t_w1, t1t, 768, 256, 768);
  launch_wconv(stream, t_w2, t2t, 256, 128, 256);
  launch_wconv(stream, q_w,  qt,  256, 512, 256);
  launch_wconv(stream, k_w,  kt,  256, 512, 256);
  launch_wconv(stream, v_w,  vt,  256, 512, 256);
  launch_wconv(stream, s_w,  st,  256, 512, 256);
  launch_wconv(stream, p_w1, p1t, 928, 1024, 928);
  launch_wconv(stream, p_w2, p2t, 1024, 256, 1024);
  launch_wconv(stream, dg_w, dgt, 256, 400, 256);
  launch_wconv(stream, dt_w, dtt, 256, 768, 256);

  // 2) GRAM encoder: LN(400) -> 512 leaky(0.2) -> 256 -> combined[:,0:256]
  layernorm_f16_kernel<<<dim3(NN), dim3(128), 0, stream>>>(gram, g_ln_g, g_ln_b, gln, 400, 416, 1.f / 400.f);
  launch_gemm(stream, gln, 416, g1t, 416, g_b1, nullptr, 0, h512, 512, NN, 512, 416, ACT_LEAKY02);
  launch_gemm(stream, h512, 512, g2t, 512, g_b2, nullptr, 0, comb, 928, NN, 256, 512, ACT_NONE);

  // 3) Text encoder: LN(768) -> 256 gelu -> 128 -> combined[:,288:416]
  layernorm_f16_kernel<<<dim3(NN), dim3(128), 0, stream>>>(text, t_ln_g, t_ln_b, tln, 768, 768, 1.f / 768.f);
  launch_gemm(stream, tln, 768, t1t, 768, t_b1, nullptr, 0, th, 256, NN, 256, 768, ACT_GELU);
  launch_gemm(stream, th, 256, t2t, 256, t_b2, nullptr, 0, comb + 288, 928, NN, 128, 256, ACT_NONE);

  // 4) Geo encoder -> combined[:,256:288]
  geo_encoder_kernel<<<dim3((NN + 255) / 256), dim3(256), 0, stream>>>(
      geo, o_ln_g, o_ln_b, o_w1, o_b1, o_w2, o_b2, comb);

  // 5) TransformerConv: q/k/v (f32) and skip (into sbuf) from x = combined[:,0:256] (f16).
  launch_gemm(stream, comb, 928, qt, 256, q_b, qb, 512, nullptr, 0, NN, 512, 256, ACT_NONE);
  launch_gemm(stream, comb, 928, kt, 256, k_b, kb, 512, nullptr, 0, NN, 512, 256, ACT_NONE);
  launch_gemm(stream, comb, 928, vt, 256, v_b, vb, 512, nullptr, 0, NN, 512, 256, ACT_NONE);
  launch_gemm(stream, comb, 928, st, 256, s_b, sbuf, 512, nullptr, 0, NN, 512, 256, ACT_NONE);

  attn_init_kernel<<<dim3((NN * 2 + 255) / 256), dim3(256), 0, stream>>>(mbuf, denom);
  {
    long waves = (long)NE * HEADS;
    long blocks = (waves * 32 + 255) / 256;
    edge_score_kernel<<<dim3((unsigned)blocks), dim3(256), 0, stream>>>(qb, kb, ei, score, mbuf);
    edge_exp_kernel<<<dim3((NE * HEADS + 255) / 256), dim3(256), 0, stream>>>(ei, mbuf, score, denom);
    edge_msg_kernel<<<dim3((unsigned)blocks), dim3(256), 0, stream>>>(vb, ei, score, denom, sbuf);
  }
  {
    long total = (long)NN * 512;
    graph_epilogue_kernel<<<dim3((unsigned)((total + 255) / 256)), dim3(256), 0, stream>>>(sbuf, comb);
  }

  // 6) Projection: combined[N,928] -> 1024 elu -> emb[N,256] (f32 out + f16 for decoders)
  launch_gemm(stream, comb, 928, p1t, 928, p_b1, nullptr, 0, h1024, 1024, NN, 1024, 928, ACT_ELU);
  launch_gemm(stream, h1024, 1024, p2t, 1024, p_b2, emb, 256, embh, 256, NN, 256, 1024, ACT_NONE);

  // 7) Decoders
  launch_gemm(stream, embh, 256, dgt, 256, dg_b, rec_gram, 400, nullptr, 0, NN, 400, 256, ACT_NONE);
  launch_gemm(stream, embh, 256, dtt, 256, dt_b, rec_text, 768, nullptr, 0, NN, 768, 256, ACT_NONE);
  geo_decoder_kernel<<<dim3((NN + 255) / 256), dim3(256), 0, stream>>>(emb, do_w, do_b, rec_geo);
}
